// Model3_75179107550044
// MI455X (gfx1250) — matrix operand order repaired and checked by reading, not compiled, not run on hardware
//
#include <hip/hip_runtime.h>
#include <math.h>

// ---------------- problem constants ----------------
#define B_     16
#define L_     96
#define NCH    321
#define MARK_  4
#define TOK    325          // NCH + MARK
#define DM     512
#define DSTATE 16
#define DTR    33
#define PREDN  96
#define ROWS   (B_*TOK)     // 5200

typedef __attribute__((ext_vector_type(16))) __bf16 v16bf;
typedef __attribute__((ext_vector_type(8)))  float  v8f;
typedef unsigned short ushort_t;

// ---------------- f32 -> bf16 (round-to-nearest-even) ----------------
static __device__ __forceinline__ ushort_t bf16_of(float x) {
  unsigned u = __float_as_uint(x);
  u = (u + 0x7FFFu + ((u >> 16) & 1u)) >> 16;
  return (ushort_t)u;
}

union FragU { v16bf v; unsigned u[8]; };

// ---- A-fragment (16x32 bf16, MxK): lane = m + 16*g.
// dword v<4 : K = 8g+2v,+1 ; dword v>=4 : K = 16+8g+2(v-4),+1.
// With bf16 storage this is two contiguous 16B loads (8 bf16 each).
static __device__ __forceinline__ v16bf load_frag_a(const ushort_t* __restrict__ rowp, int kb, int g) {
  const int k0 = kb + 8*g;
  uint4 a = *(const uint4*)(rowp + k0);        // K = k0..k0+7
  uint4 b = *(const uint4*)(rowp + k0 + 16);   // K = k0+16..k0+23
  FragU f;
  f.u[0]=a.x; f.u[1]=a.y; f.u[2]=a.z; f.u[3]=a.w;
  f.u[4]=b.x; f.u[5]=b.y; f.u[6]=b.z; f.u[7]=b.w;
  return f.v;
}

// ---- B-fragment (32x16 bf16, KxN), B[k][n]=W[n][k]: lane = n + 16*g.
// dword v<4 : K = 8g+2v,+1 ; dword v>=4 : K = 16+8g+2(v-4),+1 (same as A; two 16B loads).
static __device__ __forceinline__ v16bf load_frag_b(const ushort_t* __restrict__ rowp, int kb, int g) {
  const int k0 = kb + 8*g;
  uint4 a = *(const uint4*)(rowp + k0);        // K = k0..k0+7
  uint4 b = *(const uint4*)(rowp + k0 + 16);   // K = k0+16..k0+23
  FragU f;
  f.u[0]=a.x; f.u[1]=a.y; f.u[2]=a.z; f.u[3]=a.w;
  f.u[4]=b.x; f.u[5]=b.y; f.u[6]=b.z; f.u[7]=b.w;
  return f.v;
}

static __device__ __forceinline__ v8f wmma_bf16(v16bf a, v16bf b, v8f c) {
  return __builtin_amdgcn_wmma_f32_16x16x32_bf16(false, a, false, b, (short)0, c, false, false);
}

// ---------------- WMMA GEMM:  act(A[M,K] @ W[N,K]^T + bias) -> f32 C and/or bf16 Cb ----------------
// A, W are bf16 (rows 16B-aligned, K multiple of 32). block = 128 thr = 4 waves (2x2);
// wave tile 32x64 (8 WMMA / K-step); block tile 64x128.
// ACT: 0=none, 1=softplus, 2=gelu(exact). Cb written only for col < NBL (stride ldb).
template<int ACT>
__global__ __launch_bounds__(128)
void gemm_wmma(const ushort_t* __restrict__ A, int lda,
               const ushort_t* __restrict__ W, int ldw,
               const float* __restrict__ bias,
               float* __restrict__ C, int ldc,
               ushort_t* __restrict__ Cb, int ldb, int NBL,
               int M, int N, int K)
{
  const int lane = threadIdx.x & 31;
  const int wave = threadIdx.x >> 5;
  const int wm = wave >> 1, wn = wave & 1;
  const int g = lane >> 4, r = lane & 15;
  const int m0 = blockIdx.x*64  + wm*32;
  const int n0 = blockIdx.y*128 + wn*64;

  const ushort_t* arow[2];
#pragma unroll
  for (int i=0;i<2;i++){
    int row = m0 + i*16 + r; row = (row < M) ? row : (M-1);
    arow[i] = A + (size_t)row*lda;
  }
  const ushort_t* brow[4];
#pragma unroll
  for (int j=0;j<4;j++){
    int col = n0 + j*16 + r; col = (col < N) ? col : (N-1);
    brow[j] = W + (size_t)col*ldw;
  }

  v8f acc[2][4];
#pragma unroll
  for (int i=0;i<2;i++)
#pragma unroll
    for (int j=0;j<4;j++)
#pragma unroll
      for (int q=0;q<8;q++) acc[i][j][q]=0.f;

  for (int kb=0; kb<K; kb+=32) {
    v16bf af[2], bf[4];
#pragma unroll
    for (int i=0;i<2;i++){
      if (kb + 32 < K) __builtin_prefetch(arow[i] + kb + 32, 0, 1);  // global_prefetch_b8
      af[i] = load_frag_a(arow[i], kb, g);
    }
#pragma unroll
    for (int j=0;j<4;j++)
      bf[j] = load_frag_b(brow[j], kb, g);
#pragma unroll
    for (int i=0;i<2;i++)
#pragma unroll
      for (int j=0;j<4;j++)
        acc[i][j] = wmma_bf16(af[i], bf[j], acc[i][j]);
  }

  // C/D layout: VGPR vr, lanes0-15: (M=vr, N=lane); lanes16-31: (M=vr+8, N=lane-16)
#pragma unroll
  for (int i=0;i<2;i++)
#pragma unroll
    for (int j=0;j<4;j++) {
      int col = n0 + j*16 + r;
      if (col >= N) continue;
      float bv = bias ? bias[col] : 0.f;
#pragma unroll
      for (int vr=0; vr<8; vr++){
        int row = m0 + i*16 + vr + 8*g;
        if (row >= M) continue;
        float v = acc[i][j][vr] + bv;
        if (ACT==1) v = (v > 20.f) ? v : log1pf(__expf(v));               // softplus
        else if (ACT==2) v = 0.5f*v*(1.f + erff(v*0.70710678118f));       // exact gelu
        if (C)  C[(size_t)row*ldc + col] = v;
        if (Cb && col < NBL) Cb[(size_t)row*ldb + col] = bf16_of(v);
      }
    }
}

// ---------------- weight conversion / utility kernels ----------------
// dst[r, 0:Kdst) = bf16(src[r, 0:Ksrc)) zero-padded to Kdst
__global__ void cvt_pad_k(const float* __restrict__ src, ushort_t* __restrict__ dst,
                          int rows, int Ksrc, int Kdst) {
  int i = blockIdx.x*blockDim.x + threadIdx.x;
  if (i >= rows*Kdst) return;
  int k = i % Kdst; int r = i / Kdst;
  dst[i] = (k < Ksrc) ? bf16_of(src[(size_t)r*Ksrc + k]) : (ushort_t)0;
}

__global__ void zero16_k(ushort_t* __restrict__ p, int n) {
  int i = blockIdx.x*blockDim.x + threadIdx.x;
  if (i < n) p[i] = 0;
}

// ---------------- elementwise / small kernels ----------------
__global__ void norm_stats_k(const float* __restrict__ xe,
                             float* __restrict__ mean, float* __restrict__ stdev) {
  int i = blockIdx.x*blockDim.x + threadIdx.x;
  if (i >= B_*NCH) return;
  int b = i / NCH, c = i % NCH;
  const float* p = xe + (size_t)b*L_*NCH + c;
  float s=0.f, ss=0.f;
  for (int l=0;l<L_;l++){ float v = p[(size_t)l*NCH]; s+=v; ss+=v*v; }
  float mu = s*(1.f/(float)L_);
  float var = ss*(1.f/(float)L_) - mu*mu;
  mean[i] = mu;
  stdev[i] = sqrtf(var + 1e-5f);
}

// token matrix built directly in bf16 (only consumed by the embedding GEMM)
__global__ void build_tok_k(const float* __restrict__ xe, const float* __restrict__ xm,
                            const float* __restrict__ mean, const float* __restrict__ stdev,
                            ushort_t* __restrict__ tok) {
  int i = blockIdx.x*blockDim.x + threadIdx.x;
  if (i >= ROWS*L_) return;
  int t = i % L_; int ct = i / L_; int c = ct % TOK; int b = ct / TOK;
  float v;
  if (c < NCH) v = (xe[(size_t)b*L_*NCH + (size_t)t*NCH + c] - mean[b*NCH+c]) / stdev[b*NCH+c];
  else         v = xm[(size_t)b*L_*MARK_ + (size_t)t*MARK_ + (c-NCH)];
  tok[i] = bf16_of(v);
}

// bf16 token-flip (feeds the bwd in_proj GEMM)
__global__ void flip16_k(const ushort_t* __restrict__ in, ushort_t* __restrict__ out) {
  int i = blockIdx.x*blockDim.x + threadIdx.x;
  if (i >= ROWS*DM) return;
  int d = i % DM; int row = i / DM; int b = row / TOK; int t = row % TOK;
  out[i] = in[(size_t)(b*TOK + (TOK-1-t))*DM + d];
}

// depthwise causal conv (k=2) + SiLU; dual-writes f32 (scan) and bf16 (x_proj GEMM)
__global__ void conv_silu_k(const float* __restrict__ xz, const float* __restrict__ cw,
                            const float* __restrict__ cb,
                            float* __restrict__ xc, ushort_t* __restrict__ xcb) {
  int i = blockIdx.x*blockDim.x + threadIdx.x;
  if (i >= ROWS*DM) return;
  int d = i % DM; int row = i / DM; int t = row % TOK;
  float cur  = xz[(size_t)row*(2*DM) + d];
  float prev = (t>0) ? xz[(size_t)(row-1)*(2*DM) + d] : 0.f;
  float v = prev*cw[2*d] + cur*cw[2*d+1] + cb[d];
  float s = v / (1.f + __expf(-v));   // silu
  xc[i]  = s;
  xcb[i] = bf16_of(s);
}

// selective scan: one lane per (b,d), 16 states in registers, 325 sequential steps.
// Output written in bf16 (only consumed by out_proj GEMM).
__global__ void scan_k(const float* __restrict__ dt, const float* __restrict__ xc,
                       const float* __restrict__ xdbl, const float* __restrict__ xz,
                       const float* __restrict__ A_log, const float* __restrict__ Dp,
                       ushort_t* __restrict__ y) {
  int i = blockIdx.x*blockDim.x + threadIdx.x;
  if (i >= B_*DM) return;
  int b = i / DM, d = i % DM;
  float Ad[DSTATE];
#pragma unroll
  for (int n=0;n<DSTATE;n++) Ad[n] = -__expf(A_log[d*DSTATE+n]);
  const float Dv = Dp[d];
  float h[DSTATE];
#pragma unroll
  for (int n=0;n<DSTATE;n++) h[n]=0.f;
  for (int t=0;t<TOK;t++) {
    size_t row = (size_t)b*TOK + t;
    float dtv = dt[row*DM + d];
    float u   = xc[row*DM + d];
    const float* bc = xdbl + row*65;   // [0:33)=dt_in, [33:49)=B, [49:65)=C
    float du = dtv*u;
    float acc = 0.f;
#pragma unroll
    for (int n=0;n<DSTATE;n++) {
      float hb = h[n]*__expf(dtv*Ad[n]) + du*bc[DTR+n];
      h[n]=hb;
      acc += hb*bc[DTR+DSTATE+n];
    }
    float z = xz[row*(2*DM) + DM + d];
    y[row*DM + d] = bf16_of((acc + u*Dv) * (z / (1.f + __expf(-z))));
  }
}

// out = LayerNorm(x + a1? + a2?(optionally token-flipped)) * w + b ; dual f32/bf16 output
__global__ __launch_bounds__(128)
void add_ln_k(const float* __restrict__ x, const float* __restrict__ a1,
              const float* __restrict__ a2, int flip2,
              const float* __restrict__ w, const float* __restrict__ bsh,
              float* __restrict__ out, ushort_t* __restrict__ outb) {
  int row = blockIdx.x;
  int tid = threadIdx.x;
  int b = row / TOK, t = row % TOK;
  int row2 = flip2 ? (b*TOK + (TOK-1-t)) : row;
  float s[4];
  float lsum=0.f, lsq=0.f;
#pragma unroll
  for (int i=0;i<4;i++){
    int j = tid + i*128;
    float v = x[(size_t)row*DM + j];
    if (a1) v += a1[(size_t)row*DM + j];
    if (a2) v += a2[(size_t)row2*DM + j];
    s[i]=v; lsum+=v; lsq+=v*v;
  }
  __shared__ float red[128];
  red[tid]=lsum; __syncthreads();
  for (int o=64;o>0;o>>=1){ if (tid<o) red[tid]+=red[tid+o]; __syncthreads(); }
  float mu = red[0]*(1.f/(float)DM); __syncthreads();
  red[tid]=lsq; __syncthreads();
  for (int o=64;o>0;o>>=1){ if (tid<o) red[tid]+=red[tid+o]; __syncthreads(); }
  float var = red[0]*(1.f/(float)DM) - mu*mu;
  float inv = rsqrtf(var + 1e-5f);
#pragma unroll
  for (int i=0;i<4;i++){
    int j = tid + i*128;
    float v = (s[i]-mu)*inv*w[j] + bsh[j];
    out[(size_t)row*DM + j] = v;
    if (outb) outb[(size_t)row*DM + j] = bf16_of(v);
  }
}

// projection tail: only token NCH-1 matters
__global__ void proj_tail_k(const float* __restrict__ hn, const float* __restrict__ pw,
                            const float* __restrict__ pb, const float* __restrict__ stdev,
                            const float* __restrict__ mean, float* __restrict__ x0) {
  int i = blockIdx.x*blockDim.x + threadIdx.x;
  if (i >= B_*PREDN) return;
  int b = i / PREDN, t = i % PREDN;
  const float* hr = hn + (size_t)(b*TOK + (NCH-1))*DM;
  const float* wr = pw + (size_t)t*DM;
  float acc=0.f;
  for (int k=0;k<DM;k++) acc += hr[k]*wr[k];
  acc += pb[t];
  x0[i] = acc * stdev[b*NCH + (NCH-1)] + mean[b*NCH + (NCH-1)];
}

static __device__ __forceinline__ float geluf(float x){
  return 0.5f*x*(1.f + erff(x*0.70710678118f));
}

// head: merged->fc1->gelu ; y_t->fc3->gelu ; concat->fc2->gelu ; fc4 -> out. One block per batch.
__global__ __launch_bounds__(128)
void head_k(const float* __restrict__ x0, const float* __restrict__ x_t,
            const float* __restrict__ y_t,
            const float* __restrict__ f1w, const float* __restrict__ f1b,
            const float* __restrict__ f2w, const float* __restrict__ f2b,
            const float* __restrict__ f3w, const float* __restrict__ f3b,
            const float* __restrict__ f4w, const float* __restrict__ f4b,
            float* __restrict__ out) {
  int b = blockIdx.x; int tid = threadIdx.x;
  __shared__ float sm[768]; __shared__ float sy[384];
  __shared__ float sx1[96]; __shared__ float sy1[96]; __shared__ float sx2[512];
  for (int j=tid; j<768; j+=128) {
    int t = j >> 3, c = j & 7;
    sm[j] = (c<7) ? x_t[(size_t)b*PREDN*7 + t*7 + c] : x0[b*PREDN + t];
  }
  for (int j=tid; j<384; j+=128) sy[j] = y_t[(size_t)b*384 + j];
  __syncthreads();
  for (int o=tid; o<96; o+=128) {
    float a=0.f; const float* wr = f1w + (size_t)o*768;
    for (int k=0;k<768;k++) a += sm[k]*wr[k];
    sx1[o] = geluf(a + f1b[o]);
    float c=0.f; const float* w3 = f3w + (size_t)o*384;
    for (int k=0;k<384;k++) c += sy[k]*w3[k];
    sy1[o] = geluf(c + f3b[o]);
  }
  __syncthreads();
  for (int o=tid; o<512; o+=128) {
    float a=0.f; const float* wr = f2w + (size_t)o*192;
    for (int k=0;k<96;k++) a += sx1[k]*wr[k];
    for (int k=0;k<96;k++) a += sy1[k]*wr[96+k];
    sx2[o] = geluf(a + f2b[o]);
  }
  __syncthreads();
  for (int o=tid; o<96; o+=128) {
    float a=0.f; const float* wr = f4w + (size_t)o*DM;
    for (int k=0;k<DM;k++) a += sx2[k]*wr[k];
    out[(size_t)b*PREDN + o] = a + f4b[o];
  }
}

// ---------------- host side ----------------
struct MambaPtrs { const float *A_log,*D,*conv_b,*conv_w,*dt_b,*dt_w,*in_p,*out_p,*x_p; };
struct LayerPtrs { MambaPtrs mf, mb; const float *ln1w,*ln1b,*ln2w,*ln2b,*c1w,*c1b,*c2w,*c2b; };
struct AllPtrs {
  const float *emb_w,*emb_b,*norm_w,*norm_b,*proj_w,*proj_b;
  const float *f1w,*f1b,*f2w,*f2b,*f3w,*f3b,*f4w,*f4b;
  const float *x_enc,*x_mark,*x_t,*y_t;
  LayerPtrs L[2];
};

extern "C" void kernel_launch(void* const* d_in, const int* in_sizes, int n_in,
                              void* d_out, int out_size, void* d_ws, size_t ws_size,
                              hipStream_t stream)
{
  (void)out_size; (void)ws_size;
  auto P = [&](int i)->const float* { return (const float*)d_in[i]; };

  AllPtrs a;
  // Two plausible pytree flattening orders; pick via first leaf size.
  bool sorted_order = (n_in > 0 && in_sizes[0] == DM);   // sorted: emb_b (512) first
  if (sorted_order) {
    a.emb_b=P(0); a.emb_w=P(1);
    a.f1b=P(2); a.f1w=P(3); a.f2b=P(4); a.f2w=P(5);
    a.f3b=P(6); a.f3w=P(7); a.f4b=P(8); a.f4w=P(9);
    for (int l=0;l<2;l++){
      int lb = 10 + 26*l;
      LayerPtrs& Lp = a.L[l];
      Lp.c1b=P(lb+0); Lp.c1w=P(lb+1); Lp.c2b=P(lb+2); Lp.c2w=P(lb+3);
      Lp.ln1b=P(lb+4); Lp.ln1w=P(lb+5); Lp.ln2b=P(lb+6); Lp.ln2w=P(lb+7);
      int mb = lb+8, mf = lb+17;   // sorted mamba keys: A_log,D,conv_b,conv_w,dt_b,dt_w,in,out,x
      Lp.mb = { P(mb+0),P(mb+1),P(mb+2),P(mb+3),P(mb+4),P(mb+5),P(mb+6),P(mb+7),P(mb+8) };
      Lp.mf = { P(mf+0),P(mf+1),P(mf+2),P(mf+3),P(mf+4),P(mf+5),P(mf+6),P(mf+7),P(mf+8) };
    }
    a.norm_b=P(62); a.norm_w=P(63); a.proj_b=P(64); a.proj_w=P(65);
    a.x_enc=P(66); a.x_mark=P(67); a.x_t=P(70); a.y_t=P(71);
  } else {
    // insertion order; mamba leaves: 0 in_proj,1 conv_w,2 conv_b,3 x_proj,
    // 4 dt_proj_w,5 dt_proj_b,6 A_log,7 D,8 out_proj
    a.emb_w=P(0); a.emb_b=P(1);
    for (int l=0;l<2;l++){
      int lb = 2 + 26*l;
      LayerPtrs& Lp = a.L[l];
      int mf = lb+0, mb = lb+9;
      Lp.mf = { P(mf+6),P(mf+7),P(mf+2),P(mf+1),P(mf+5),P(mf+4),P(mf+0),P(mf+8),P(mf+3) };
      Lp.mb = { P(mb+6),P(mb+7),P(mb+2),P(mb+1),P(mb+5),P(mb+4),P(mb+0),P(mb+8),P(mb+3) };
      Lp.ln1w=P(lb+18); Lp.ln1b=P(lb+19);
      Lp.c1w=P(lb+20); Lp.c1b=P(lb+21); Lp.c2w=P(lb+22); Lp.c2b=P(lb+23);
      Lp.ln2w=P(lb+24); Lp.ln2b=P(lb+25);
    }
    a.norm_w=P(54); a.norm_b=P(55); a.proj_w=P(56); a.proj_b=P(57);
    a.f1w=P(58); a.f1b=P(59); a.f2w=P(60); a.f2b=P(61);
    a.f3w=P(62); a.f3b=P(63); a.f4w=P(64); a.f4b=P(65);
    a.x_enc=P(66); a.x_mark=P(67); a.x_t=P(70); a.y_t=P(71);
  }

  // ---------------- workspace layout ----------------
  // f32 region first, then bf16 (ushort) region; all offsets keep 16B alignment.
  float* wf = (float*)d_ws;
  size_t o = 0;
  float* means = wf+o; o += (size_t)B_*NCH;
  float* stdev = wf+o; o += (size_t)B_*NCH;
  float* h     = wf+o; o += (size_t)ROWS*DM;
  float* h2    = wf+o; o += (size_t)ROWS*DM;
  float* xz    = wf+o; o += (size_t)ROWS*2*DM;
  float* xc    = wf+o; o += (size_t)ROWS*DM;     // also FFN c2 output
  float* xdbl  = wf+o; o += (size_t)ROWS*65;
  float* dtb   = wf+o; o += (size_t)ROWS*DM;
  float* nxf   = wf+o; o += (size_t)ROWS*DM;
  float* nxb   = wf+o; o += (size_t)ROWS*DM;
  float* x0    = wf+o; o += (size_t)B_*PREDN;

  ushort_t* wu = (ushort_t*)(wf + o);
  size_t u = 0;
  ushort_t* h_bf    = wu+u; u += (size_t)ROWS*DM;
  ushort_t* h2_bf   = wu+u; u += (size_t)ROWS*DM;
  ushort_t* xc_bf   = wu+u; u += (size_t)ROWS*DM;
  ushort_t* yb_bf   = wu+u; u += (size_t)ROWS*DM;      // scan output; tok_bf aliases it
  ushort_t* ffn_bf  = wu+u; u += (size_t)ROWS*DM;      // c1 output; xf_bf aliases it
  ushort_t* xdbl_bf = wu+u; u += (size_t)ROWS*64;      // zero-padded K for dt GEMM
  ushort_t* emb_wb  = wu+u; u += (size_t)DM*L_;
  ushort_t* c1_wb[2], *c2_wb[2];
  for (int l=0;l<2;l++){ c1_wb[l]=wu+u; u+=(size_t)DM*DM; c2_wb[l]=wu+u; u+=(size_t)DM*DM; }
  ushort_t* in_wb[2][2], *x_wb[2][2], *dt_wb[2][2], *out_wb[2][2];
  for (int l=0;l<2;l++) for (int d=0;d<2;d++){
    in_wb[l][d]=wu+u;  u+=(size_t)2*DM*DM;
    x_wb[l][d]=wu+u;   u+=(size_t)65*DM + 8;   // +8 keeps next offset 16B-aligned
    dt_wb[l][d]=wu+u;  u+=(size_t)DM*64;
    out_wb[l][d]=wu+u; u+=(size_t)DM*DM;
  }
  ushort_t* tok_bf = yb_bf;   // token matrix only lives until emb GEMM; scan runs later
  ushort_t* xf_bf  = ffn_bf;  // flipped input only lives until bwd in_proj; c1 runs later

  const dim3 blk(128);
  const int gM = (ROWS + 63)/64;     // 82
  const int EW = 256;
  auto g1 = [&](int n){ return dim3((n + EW - 1)/EW); };

  // -------- weight conversion (once per launch; amortized over 82 M-blocks) --------
  cvt_pad_k<<<g1(DM*L_), dim3(EW), 0, stream>>>(a.emb_w, emb_wb, DM, L_, L_);
  for (int l=0;l<2;l++){
    const LayerPtrs& Lp = a.L[l];
    cvt_pad_k<<<g1(DM*DM), dim3(EW), 0, stream>>>(Lp.c1w, c1_wb[l], DM, DM, DM);
    cvt_pad_k<<<g1(DM*DM), dim3(EW), 0, stream>>>(Lp.c2w, c2_wb[l], DM, DM, DM);
    for (int d=0;d<2;d++){
      const MambaPtrs& m = d ? Lp.mb : Lp.mf;
      cvt_pad_k<<<g1(2*DM*DM), dim3(EW), 0, stream>>>(m.in_p,  in_wb[l][d],  2*DM, DM, DM);
      cvt_pad_k<<<g1(65*DM),   dim3(EW), 0, stream>>>(m.x_p,   x_wb[l][d],   65, DM, DM);
      cvt_pad_k<<<g1(DM*64),   dim3(EW), 0, stream>>>(m.dt_w,  dt_wb[l][d],  DM, DTR, 64);
      cvt_pad_k<<<g1(DM*DM),   dim3(EW), 0, stream>>>(m.out_p, out_wb[l][d], DM, DM, DM);
    }
  }
  zero16_k<<<g1(ROWS*64), dim3(EW), 0, stream>>>(xdbl_bf, ROWS*64);  // pad cols 33..63 = 0

  auto run_mamba = [&](const ushort_t* in_bf, const MambaPtrs& m,
                       const ushort_t* in_w, const ushort_t* x_w,
                       const ushort_t* dt_w, const ushort_t* out_w, float* nx) {
    // xz = in @ in_proj^T              (5200 x 1024, K=512)
    gemm_wmma<0><<<dim3(gM,(2*DM+127)/128), blk, 0, stream>>>(
        in_bf, DM, in_w, DM, nullptr, xz, 2*DM, nullptr, 0, 0, ROWS, 2*DM, DM);
    // depthwise conv + silu (f32 + bf16)
    conv_silu_k<<<g1(ROWS*DM), dim3(EW), 0, stream>>>(xz, m.conv_w, m.conv_b, xc, xc_bf);
    // xdbl = xc @ x_proj^T             (5200 x 65, K=512); bf16 side-copy of cols<33 (ld 64)
    gemm_wmma<0><<<dim3(gM,1), blk, 0, stream>>>(
        xc_bf, DM, x_w, DM, nullptr, xdbl, 65, xdbl_bf, 64, DTR, ROWS, 65, DM);
    // dt = softplus(xdbl[:, :33] @ dt_proj^T + b)   zero-padded K=64 -> branchless
    gemm_wmma<1><<<dim3(gM,(DM+127)/128), blk, 0, stream>>>(
        xdbl_bf, 64, dt_w, 64, m.dt_b, dtb, DM, nullptr, 0, 0, ROWS, DM, 64);
    // selective scan + D skip + silu(z) gate -> bf16
    scan_k<<<g1(B_*DM), dim3(EW), 0, stream>>>(dtb, xc, xdbl, xz, m.A_log, m.D, yb_bf);
    // nx = y @ out_proj^T              (5200 x 512, K=512)
    gemm_wmma<0><<<dim3(gM,(DM+127)/128), blk, 0, stream>>>(
        yb_bf, DM, out_w, DM, nullptr, nx, DM, nullptr, 0, 0, ROWS, DM, DM);
  };

  // -------- instance norm stats + token embedding --------
  norm_stats_k<<<g1(B_*NCH), dim3(EW), 0, stream>>>(a.x_enc, means, stdev);
  build_tok_k<<<g1(ROWS*L_), dim3(EW), 0, stream>>>(a.x_enc, a.x_mark, means, stdev, tok_bf);
  gemm_wmma<0><<<dim3(gM,(DM+127)/128), blk, 0, stream>>>(
      tok_bf, L_, emb_wb, L_, a.emb_b, h, DM, h_bf, DM, DM, ROWS, DM, L_);

  // -------- encoder layers --------
  for (int l=0; l<2; l++) {
    const LayerPtrs& Lp = a.L[l];
    run_mamba(h_bf, Lp.mf, in_wb[l][0], x_wb[l][0], dt_wb[l][0], out_wb[l][0], nxf);
    flip16_k<<<g1(ROWS*DM), dim3(EW), 0, stream>>>(h_bf, xf_bf);
    run_mamba(xf_bf, Lp.mb, in_wb[l][1], x_wb[l][1], dt_wb[l][1], out_wb[l][1], nxb);
    add_ln_k<<<dim3(ROWS), blk, 0, stream>>>(h, nxf, nxb, 1, Lp.ln1w, Lp.ln1b, h2, h2_bf);
    // FFN: c1 (gelu) -> bf16 only ; c2 -> f32 ; LN2 -> h (f32 + bf16)
    gemm_wmma<2><<<dim3(gM,(DM+127)/128), blk, 0, stream>>>(
        h2_bf, DM, c1_wb[l], DM, Lp.c1b, nullptr, 0, ffn_bf, DM, DM, ROWS, DM, DM);
    gemm_wmma<0><<<dim3(gM,(DM+127)/128), blk, 0, stream>>>(
        ffn_bf, DM, c2_wb[l], DM, Lp.c2b, xc, DM, nullptr, 0, 0, ROWS, DM, DM);
    add_ln_k<<<dim3(ROWS), blk, 0, stream>>>(h2, xc, nullptr, 0, Lp.ln2w, Lp.ln2b, h, h_bf);
  }

  // -------- final norm, projection tail, head --------
  add_ln_k<<<dim3(ROWS), blk, 0, stream>>>(h, nullptr, nullptr, 0, a.norm_w, a.norm_b, h2, nullptr);
  proj_tail_k<<<g1(B_*PREDN), dim3(EW), 0, stream>>>(h2, a.proj_w, a.proj_b, stdev, means, x0);
  head_k<<<dim3(B_), blk, 0, stream>>>(x0, a.x_t, a.y_t,
                                       a.f1w, a.f1b, a.f2w, a.f2b,
                                       a.f3w, a.f3b, a.f4w, a.f4b,
                                       (float*)d_out);
}
